// SelfDecoderBlock_39393440039480
// MI455X (gfx1250) — compile-verified, matrix operand order checked by reading, not run on hardware
//
#include <hip/hip_runtime.h>
#include <math.h>

// ---------------------------------------------------------------------------
// MI455X (gfx1250) decoder block: LN+RoPE -> QKV (WMMA bf16 GEMM) ->
// flash attention (WMMA) -> Wo GEMM -> LN -> GeGLU FFN (WMMA GEMMs).
// Compute-bound (~585 GFLOP vs ~0.4 GB traffic) => all matmuls on
// v_wmma_f32_16x16x32_bf16 with fp32 accumulate; GEMM tiles staged into LDS
// with the gfx1250 async-to-LDS path (ASYNCcnt) and double buffering.
// ---------------------------------------------------------------------------

typedef __attribute__((ext_vector_type(16))) __bf16        v16bf;
typedef __attribute__((ext_vector_type(8)))  float         v8f;
typedef __attribute__((ext_vector_type(4)))  unsigned int  v4u;
typedef __attribute__((ext_vector_type(4)))  int           v4i;

#define B_  4
#define S_  1024
#define E_  2048
#define H_  16
#define DH_ 128
#define F_  8192
#define NB_ (B_ * S_)     // 4096 rows
#define HD_ (H_ * DH_)    // 2048

#if __has_builtin(__builtin_amdgcn_global_load_async_to_lds_b128) && \
    __has_builtin(__builtin_amdgcn_s_wait_asynccnt)
#define USE_ASYNC_LDS 1
#else
#define USE_ASYNC_LDS 0
#endif

#if USE_ASYNC_LDS
typedef __attribute__((address_space(1))) v4i as1_v4i;   // global int4
typedef __attribute__((address_space(3))) v4i as3_v4i;   // LDS int4
#endif

// ------------------------- bf16 helpers (explicit bit ops) ------------------
__device__ __forceinline__ unsigned short f2bf(float f) {
  unsigned int u = __builtin_bit_cast(unsigned int, f);
  unsigned int r = u + 0x7FFFu + ((u >> 16) & 1u);   // round to nearest even
  return (unsigned short)(r >> 16);
}
__device__ __forceinline__ float bf2f(unsigned short h) {
  unsigned int u = ((unsigned int)h) << 16;
  return __builtin_bit_cast(float, u);
}
__device__ __forceinline__ __bf16 us2bf(unsigned short h) {
  return __builtin_bit_cast(__bf16, h);
}

// K index of pair p (covers k0, k0+1) for the 16-bit A-matrix layout:
// lanes 0-15: v0..3 -> K 0..7, v4..7 -> K 16..23 ; lanes 16-31: +8
__device__ __forceinline__ int amap_k(int pair, int group) {
  int base = (pair < 4) ? (2 * pair) : (16 + 2 * (pair - 4));
  return base + 8 * group;
}

// 16B global -> LDS transfer: async DMA on gfx1250 toolchains that expose it.
__device__ __forceinline__ void stage_b128(const void* g, void* l) {
#if USE_ASYNC_LDS
  __builtin_amdgcn_global_load_async_to_lds_b128(
      (as1_v4i*)(v4i*)(void*)g, (as3_v4i*)(v4i*)l, 0, 0);
#else
  *(v4u*)l = *(const v4u*)g;
#endif
}

// ------------------------- weight fp32 -> bf16 ------------------------------
__global__ __launch_bounds__(256) void cvt_bf16_kernel(
    const float* __restrict__ in, unsigned short* __restrict__ out, long n) {
  long i = (long)blockIdx.x * blockDim.x + threadIdx.x;
  if (i < n) out[i] = f2bf(in[i]);
}

// ------------------------- LayerNorm (+ optional RoPE) ----------------------
__global__ __launch_bounds__(256) void ln_rope_kernel(
    const float* __restrict__ X, const float* __restrict__ gamma,
    const float* __restrict__ beta, unsigned short* __restrict__ Y,
    unsigned short* __restrict__ YP, int do_rope) {
  __shared__ float xs[E_];
  __shared__ float ssum, ssq;
  int row = blockIdx.x;
  int tid = threadIdx.x;
  if (tid == 0) { ssum = 0.f; ssq = 0.f; }
  __syncthreads();

  const float* xr = X + (size_t)row * E_;
  float s = 0.f, s2 = 0.f;
  for (int j = tid; j < E_; j += 256) {
    float v = xr[j];
    xs[j] = v;
    s += v;
    s2 += v * v;
  }
#pragma unroll
  for (int msk = 16; msk >= 1; msk >>= 1) {
    s  += __shfl_xor(s,  msk, 32);
    s2 += __shfl_xor(s2, msk, 32);
  }
  if ((tid & 31) == 0) {           // one ds_add_f32 per wave
    atomicAdd(&ssum, s);
    atomicAdd(&ssq,  s2);
  }
  __syncthreads();

  float mean = ssum * (1.0f / E_);
  float var  = ssq * (1.0f / E_) - mean * mean;
  float rs   = rsqrtf(var + 1e-6f);

  const int half = E_ >> 1;
  int pos = row % S_;
  for (int j = tid; j < E_; j += 256) {
    float yv = (xs[j] - mean) * rs * gamma[j] + beta[j];
    Y[(size_t)row * E_ + j] = f2bf(yv);
    if (do_rope) {
      int fi = (j < half) ? j : j - half;
      int pj = (j < half) ? j + half : j - half;
      float yo  = (xs[pj] - mean) * rs * gamma[pj] + beta[pj];
      float rot = (j < half) ? -yo : yo;
      float invf = __expf((float)fi * (-9.210340371976184f / (float)half));
      float ang  = (float)pos * invf;
      float c, sn;
      __sincosf(ang, &sn, &c);   // (x, &sin, &cos)
      YP[(size_t)row * E_ + j] = f2bf(yv * c + rot * sn);
    }
  }
}

// ------------------------- tiled bf16 WMMA GEMM (async double-buffered) -----
// C[M,N] = A[M,K] x B[K,N], A,B bf16 row-major. 128x128 block tile, K-step 32.
// 8 waves: 4(M) x 2(N); each wave = 32x64 = 2x4 wmma tiles.
// Each thread stages exactly 4x16B per tile -> s_wait_asynccnt(4) overlaps the
// next tile's DMA with the current tile's WMMA work (async loads retire
// in order per ISA §10.7/§4.1).
// Epilogue: 0=bf16  1=bf16*alpha  2=f32  3=gelu->bf16  4=(val*aux)->bf16
template <int EPI>
__global__ __launch_bounds__(256) void gemm_bf16_kernel(
    const unsigned short* __restrict__ A, const unsigned short* __restrict__ Bm,
    void* __restrict__ Cout, const unsigned short* __restrict__ Aux,
    int M, int N, int K, float alpha) {
  __shared__ unsigned short As[2][128][40];   // 32 + 8 pad (80B rows, 16B ok)
  __shared__ unsigned short Bs[2][32][136];   // 128 + 8 pad (272B rows)

  const int tid  = threadIdx.x;
  const int lane = tid & 31;
  const int wave = tid >> 5;
  const int l16  = lane & 15;
  const int grp  = lane >> 4;
  const int wm   = wave >> 1;   // 0..3 -> +32*wm
  const int wn   = wave & 1;    // 0..1 -> +64*wn
  const int bm   = blockIdx.y * 128;
  const int bn   = blockIdx.x * 128;

  v8f acc[2][4] = {};

  const int ar = tid >> 1, ac = (tid & 1) << 4;   // A loader coords (128x32)
  const int br = tid >> 3, bc = (tid & 7) << 4;   // B loader coords (32x128)

  auto stage_tile = [&](int buf, int k0) {
    const unsigned short* asrc = A + (size_t)(bm + ar) * K + k0 + ac;
    const unsigned short* bsrc = Bm + (size_t)(k0 + br) * N + bn + bc;
    stage_b128(asrc,     &As[buf][ar][ac]);
    stage_b128(asrc + 8, &As[buf][ar][ac + 8]);
    stage_b128(bsrc,     &Bs[buf][br][bc]);
    stage_b128(bsrc + 8, &Bs[buf][br][bc + 8]);
  };

  const int ntiles = K >> 5;
  stage_tile(0, 0);
  int buf = 0;
  for (int kt = 0; kt < ntiles; ++kt) {
    if (kt + 1 < ntiles) {
      stage_tile(buf ^ 1, (kt + 1) << 5);   // prefetch next tile (4 asyncs)
#if USE_ASYNC_LDS
      __builtin_amdgcn_s_wait_asynccnt(4);  // current tile's 4 are done
#endif
    } else {
#if USE_ASYNC_LDS
      __builtin_amdgcn_s_wait_asynccnt(0);
#endif
    }
    __syncthreads();   // everyone's fills for `buf` visible

    v16bf afrag[2], bfrag[4];
#pragma unroll
    for (int i = 0; i < 2; i++) {
      int rrow = wm * 32 + i * 16 + l16;
#pragma unroll
      for (int p = 0; p < 8; p++) {
        int kk = amap_k(p, grp);
        unsigned int w = *(const unsigned int*)&As[buf][rrow][kk];
        afrag[i][2 * p]     = us2bf((unsigned short)(w & 0xFFFFu));
        afrag[i][2 * p + 1] = us2bf((unsigned short)(w >> 16));
      }
    }
#pragma unroll
    for (int j = 0; j < 4; j++) {
      int col = wn * 64 + j * 16 + l16;
#pragma unroll
      for (int p = 0; p < 8; p++) {
        int kk = amap_k(p, grp);
        bfrag[j][2 * p]     = us2bf(Bs[buf][kk][col]);
        bfrag[j][2 * p + 1] = us2bf(Bs[buf][kk + 1][col]);
      }
    }
#pragma unroll
    for (int i = 0; i < 2; i++)
#pragma unroll
      for (int j = 0; j < 4; j++)
        acc[i][j] = __builtin_amdgcn_wmma_f32_16x16x32_bf16(
            false, afrag[i], false, bfrag[j], (short)0, acc[i][j], false, false);

    __syncthreads();   // done reading `buf` before it is refilled next round
    buf ^= 1;
  }

  // epilogue
#pragma unroll
  for (int i = 0; i < 2; i++) {
#pragma unroll
    for (int j = 0; j < 4; j++) {
      int n = bn + wn * 64 + j * 16 + l16;
#pragma unroll
      for (int r = 0; r < 8; r++) {
        int m = bm + wm * 32 + i * 16 + r + 8 * grp;
        size_t idx = (size_t)m * N + n;
        float o = acc[i][j][r];
        if (EPI == 2) {
          ((float*)Cout)[idx] = o;
        } else {
          if (EPI == 1) o *= alpha;
          if (EPI == 3) o = 0.5f * o * (1.0f + erff(o * 0.70710678118654752f));
          if (EPI == 4) o = o * bf2f(Aux[idx]);
          ((unsigned short*)Cout)[idx] = f2bf(o);
        }
      }
    }
  }
}

// ------------------------- flash attention (WMMA, causal) -------------------
// one wave per (b, h, 16-query tile); online softmax over 32-key steps.
__global__ __launch_bounds__(32) void attn_kernel(
    const unsigned short* __restrict__ Q, const unsigned short* __restrict__ Kd,
    const unsigned short* __restrict__ V, unsigned short* __restrict__ O) {
  const int qt   = blockIdx.x;
  const int h    = blockIdx.y;
  const int b    = blockIdx.z;
  const int lane = threadIdx.x;
  const int l16  = lane & 15;
  const int grp  = lane >> 4;

  __shared__ unsigned short p_lds[16][40];

  // Q fragments: 16 rows x 128 d (4 k-chunks of 32)
  v16bf qfrag[4];
  {
    int qrow = qt * 16 + l16;
    const unsigned short* qb = Q + ((size_t)(b * S_ + qrow)) * HD_ + h * DH_;
#pragma unroll
    for (int kd = 0; kd < 4; kd++) {
#pragma unroll
      for (int p = 0; p < 8; p++) {
        int dk = kd * 32 + amap_k(p, grp);
        unsigned int w = *(const unsigned int*)(qb + dk);
        qfrag[kd][2 * p]     = us2bf((unsigned short)(w & 0xFFFFu));
        qfrag[kd][2 * p + 1] = us2bf((unsigned short)(w >> 16));
      }
    }
  }

  v8f o_acc[8] = {};
  float mrow[8], lrow[8];
#pragma unroll
  for (int r = 0; r < 8; r++) { mrow[r] = -3.0e38f; lrow[r] = 0.f; }

  const int qmax   = qt * 16 + 15;
  const int nsteps = (qmax >> 5) + 1;

  for (int kt = 0; kt < nsteps; kt++) {
    const int kbase = kt * 32;

    // scores = Q x K^T  (two 16-key column tiles)
    v8f sfrag[2] = {};
#pragma unroll
    for (int nf = 0; nf < 2; nf++) {
      int krow = kbase + nf * 16 + l16;
      const unsigned short* kb = Kd + ((size_t)(b * S_ + krow)) * HD_ + h * DH_;
#pragma unroll
      for (int kd = 0; kd < 4; kd++) {
        v16bf kf;
#pragma unroll
        for (int p = 0; p < 8; p++) {
          int dk = kd * 32 + amap_k(p, grp);
          unsigned int w = *(const unsigned int*)(kb + dk);
          kf[2 * p]     = us2bf((unsigned short)(w & 0xFFFFu));
          kf[2 * p + 1] = us2bf((unsigned short)(w >> 16));
        }
        sfrag[nf] = __builtin_amdgcn_wmma_f32_16x16x32_bf16(
            false, qfrag[kd], false, kf, (short)0, sfrag[nf], false, false);
      }
    }

    // causal mask + online softmax (per accumulator row)
#pragma unroll
    for (int r = 0; r < 8; r++) {
      int mg = qt * 16 + r + 8 * grp;
      float s0 = sfrag[0][r], s1 = sfrag[1][r];
      if (kbase + l16 > mg)       s0 = -3.0e38f;
      if (kbase + 16 + l16 > mg)  s1 = -3.0e38f;
      float mx = fmaxf(s0, s1);
#pragma unroll
      for (int msk = 1; msk < 16; msk <<= 1)
        mx = fmaxf(mx, __shfl_xor(mx, msk, 32));
      float mnew  = fmaxf(mrow[r], mx);
      float scale = __expf(mrow[r] - mnew);
      float e0 = __expf(s0 - mnew), e1 = __expf(s1 - mnew);
      float ps = e0 + e1;
#pragma unroll
      for (int msk = 1; msk < 16; msk <<= 1)
        ps += __shfl_xor(ps, msk, 32);
      lrow[r] = lrow[r] * scale + ps;
      mrow[r] = mnew;
#pragma unroll
      for (int dt = 0; dt < 8; dt++) o_acc[dt][r] *= scale;
      sfrag[0][r] = e0;
      sfrag[1][r] = e1;
    }

    // relayout P: C-layout f32 -> A-layout bf16 via LDS
#pragma unroll
    for (int nf = 0; nf < 2; nf++)
#pragma unroll
      for (int r = 0; r < 8; r++)
        p_lds[r + 8 * grp][nf * 16 + l16] = f2bf(sfrag[nf][r]);
    __syncthreads();
    v16bf pfrag;
#pragma unroll
    for (int p = 0; p < 8; p++) {
      int kk = amap_k(p, grp);
      unsigned int w = *(const unsigned int*)&p_lds[l16][kk];
      pfrag[2 * p]     = us2bf((unsigned short)(w & 0xFFFFu));
      pfrag[2 * p + 1] = us2bf((unsigned short)(w >> 16));
    }
    __syncthreads();

    // ctx += P x V  (8 d-tiles of 16)
#pragma unroll
    for (int dt = 0; dt < 8; dt++) {
      v16bf vf;
#pragma unroll
      for (int p = 0; p < 8; p++) {
        int key = kbase + amap_k(p, grp);
        const unsigned short* vb =
            V + ((size_t)(b * S_ + key)) * HD_ + h * DH_ + dt * 16 + l16;
        vf[2 * p]     = us2bf(vb[0]);
        vf[2 * p + 1] = us2bf(vb[HD_]);
      }
      o_acc[dt] = __builtin_amdgcn_wmma_f32_16x16x32_bf16(
          false, pfrag, false, vf, (short)0, o_acc[dt], false, false);
    }
  }

  // normalize + store ctx (bf16)
#pragma unroll
  for (int r = 0; r < 8; r++) {
    float inv = 1.0f / lrow[r];
    int mg = qt * 16 + r + 8 * grp;
    size_t orow = ((size_t)(b * S_ + mg)) * HD_ + h * DH_;
#pragma unroll
    for (int dt = 0; dt < 8; dt++)
      O[orow + dt * 16 + l16] = f2bf(o_acc[dt][r] * inv);
  }
}

// ------------------------- host-side orchestration --------------------------
extern "C" void kernel_launch(void* const* d_in, const int* in_sizes, int n_in,
                              void* d_out, int out_size, void* d_ws, size_t ws_size,
                              hipStream_t stream) {
  (void)in_sizes; (void)n_in; (void)out_size; (void)ws_size;
  const float* x      = (const float*)d_in[0];
  const float* gamma1 = (const float*)d_in[1];
  const float* beta1  = (const float*)d_in[2];
  const float* Wq     = (const float*)d_in[3];
  const float* Wk     = (const float*)d_in[4];
  const float* Wv     = (const float*)d_in[5];
  const float* Wo     = (const float*)d_in[6];
  const float* gamma2 = (const float*)d_in[7];
  const float* beta2  = (const float*)d_in[8];
  const float* Wg     = (const float*)d_in[9];
  const float* Wu     = (const float*)d_in[10];
  const float* Wd     = (const float*)d_in[11];

  char* ws = (char*)d_ws;
  size_t off = 0;
  auto take = [&](size_t bytes) -> char* {
    char* p = ws + off;
    off += (bytes + 255) & ~(size_t)255;
    return p;
  };
  const size_t rowsE = (size_t)NB_ * E_;   // 8.39M
  const size_t rowsF = (size_t)NB_ * F_;   // 33.6M
  unsigned short* y    = (unsigned short*)take(rowsE * 2);
  unsigned short* yp   = (unsigned short*)take(rowsE * 2);
  unsigned short* q    = (unsigned short*)take(rowsE * 2);
  unsigned short* k    = (unsigned short*)take(rowsE * 2);
  unsigned short* v    = (unsigned short*)take(rowsE * 2);
  unsigned short* ctx  = (unsigned short*)take(rowsE * 2);
  unsigned short* z    = (unsigned short*)take(rowsE * 2);
  unsigned short* Wq_b = (unsigned short*)take((size_t)E_ * HD_ * 2);
  unsigned short* Wk_b = (unsigned short*)take((size_t)E_ * HD_ * 2);
  unsigned short* Wv_b = (unsigned short*)take((size_t)E_ * HD_ * 2);
  unsigned short* Wo_b = (unsigned short*)take((size_t)HD_ * E_ * 2);
  unsigned short* Wg_b = (unsigned short*)take((size_t)E_ * F_ * 2);
  unsigned short* Wu_b = (unsigned short*)take((size_t)E_ * F_ * 2);
  unsigned short* Wd_b = (unsigned short*)take((size_t)F_ * E_ * 2);
  unsigned short* gate = (unsigned short*)take(rowsF * 2);
  unsigned short* hb   = (unsigned short*)take(rowsF * 2);
  float*          attn = (float*)take(rowsE * 4);

  auto cvt = [&](const float* src, unsigned short* dst, long n) {
    cvt_bf16_kernel<<<(unsigned)((n + 255) / 256), 256, 0, stream>>>(src, dst, n);
  };
  cvt(Wq, Wq_b, (long)E_ * HD_);
  cvt(Wk, Wk_b, (long)E_ * HD_);
  cvt(Wv, Wv_b, (long)E_ * HD_);
  cvt(Wo, Wo_b, (long)HD_ * E_);
  cvt(Wg, Wg_b, (long)E_ * F_);
  cvt(Wu, Wu_b, (long)E_ * F_);
  cvt(Wd, Wd_b, (long)F_ * E_);

  // LN1 + RoPE
  ln_rope_kernel<<<NB_, 256, 0, stream>>>(x, gamma1, beta1, y, yp, 1);

  const float qscale = 0.08838834764831845f;  // DH^-0.5
  dim3 blk(256);
  dim3 gE(HD_ / 128, NB_ / 128);   // N=2048
  dim3 gF(F_ / 128, NB_ / 128);    // N=8192

  // q = (yp x Wq) * qscale ; k = yp x Wk ; v = y x Wv
  gemm_bf16_kernel<1><<<gE, blk, 0, stream>>>(yp, Wq_b, q, nullptr, NB_, HD_, E_, qscale);
  gemm_bf16_kernel<0><<<gE, blk, 0, stream>>>(yp, Wk_b, k, nullptr, NB_, HD_, E_, 1.f);
  gemm_bf16_kernel<0><<<gE, blk, 0, stream>>>(y,  Wv_b, v, nullptr, NB_, HD_, E_, 1.f);

  // flash attention
  attn_kernel<<<dim3(S_ / 16, H_, B_), 32, 0, stream>>>(q, k, v, ctx);

  // attn_out = ctx x Wo (f32)
  gemm_bf16_kernel<2><<<gE, blk, 0, stream>>>(ctx, Wo_b, attn, nullptr, NB_, E_, HD_, 1.f);

  // LN2 (no rope)
  ln_rope_kernel<<<NB_, 256, 0, stream>>>(attn, gamma2, beta2, z, z, 0);

  // gate = gelu(z x Wg) ; hb = (z x Wu) * gate ; out = hb x Wd
  gemm_bf16_kernel<3><<<gF, blk, 0, stream>>>(z, Wg_b, gate, nullptr, NB_, F_, E_, 1.f);
  gemm_bf16_kernel<4><<<gF, blk, 0, stream>>>(z, Wu_b, hb, gate, NB_, F_, E_, 1.f);
  gemm_bf16_kernel<2><<<gE, blk, 0, stream>>>(hb, Wd_b, d_out, nullptr, NB_, E_, F_, 1.f);
}
